// GlobalAggregationBlock_57140244906430
// MI455X (gfx1250) — matrix operand order repaired and checked by reading, not compiled, not run on hardware
//
#include <hip/hip_runtime.h>
#include <hip/hip_bf16.h>

// ---------------------------------------------------------------------------
// Problem constants (from reference): B=1, C=128, D=16,H=32,W=32 -> N=16384,
// ck=cq=cv=64, co=128.
// ---------------------------------------------------------------------------
#define NTOK   16384
#define CIN    128
#define DK     64
#define COUT   128
#define LOG2E  1.4426950408889634f

typedef __attribute__((ext_vector_type(16))) __bf16 bf16x16;
typedef __attribute__((ext_vector_type(8)))  float  f32x8;
typedef int v4i_t __attribute__((vector_size(16)));   // matches builtin pointee

#if __has_builtin(__builtin_amdgcn_global_load_async_to_lds_b128)
#define USE_ASYNC 1
#else
#define USE_ASYNC 0
#endif

__device__ __forceinline__ f32x8 wmma_bf16(bf16x16 a, bf16x16 b, f32x8 c) {
    // (neg_a, A, neg_b, B, c_mod, C, reuse_a, reuse_b)
    return __builtin_amdgcn_wmma_f32_16x16x32_bf16(false, a, false, b,
                                                   (short)0, c, false, false);
}

#if USE_ASYNC
__device__ __forceinline__ void g2l_b128(const __bf16* g, __bf16* l) {
    // Per-lane 16B async DMA: LDS[l] = MEM[g]; tracked with ASYNCcnt.
    __builtin_amdgcn_global_load_async_to_lds_b128(
        (__attribute__((address_space(1))) v4i_t*)(g),
        (__attribute__((address_space(3))) v4i_t*)(l), 0, 0);
}
__device__ __forceinline__ void wait_async0() {
#if __has_builtin(__builtin_amdgcn_s_wait_asynccnt)
    __builtin_amdgcn_s_wait_asynccnt(0);
#else
    asm volatile("s_wait_asynccnt 0x0" ::: "memory");
#endif
}
#endif

// ---- DPP16 butterfly reduction over each 16-lane half (pure VALU, no LDS) --
template <int CTRL>
__device__ __forceinline__ float dpp_max_step(float v) {
    int t = __builtin_amdgcn_update_dpp(0, __float_as_int(v), CTRL, 0xF, 0xF, true);
    return fmaxf(v, __int_as_float(t));
}
template <int CTRL>
__device__ __forceinline__ float dpp_add_step(float v) {
    int t = __builtin_amdgcn_update_dpp(0, __float_as_int(v), CTRL, 0xF, 0xF, true);
    return v + __int_as_float(t);
}
__device__ __forceinline__ float reduce16_max(float v) {
    v = dpp_max_step<0xB1>(v);   // quad_perm xor 1
    v = dpp_max_step<0x4E>(v);   // quad_perm xor 2
    v = dpp_max_step<0x141>(v);  // row_half_mirror (lane^7 within 8)
    v = dpp_max_step<0x140>(v);  // row_mirror      (lane^15 within 16)
    return v;
}
__device__ __forceinline__ float reduce16_add(float v) {
    v = dpp_add_step<0xB1>(v);
    v = dpp_add_step<0x4E>(v);
    v = dpp_add_step<0x141>(v);
    v = dpp_add_step<0x140>(v);
    return v;
}

// ---------------------------------------------------------------------------
// Kernel 1: Q/K/V pointwise projections (fp32 compute, bf16 output).
//   Qt[n][d], Kt[n][d]  (token-major, row stride 64)
//   Vm[d][n]            (channel-major, row stride N) -> contiguous PV B-frags
// ---------------------------------------------------------------------------
__global__ __launch_bounds__(256)
void qkv_proj(const float* __restrict__ X,
              const float* __restrict__ Wq, const float* __restrict__ bq,
              const float* __restrict__ Wk, const float* __restrict__ bk,
              const float* __restrict__ Wv, const float* __restrict__ bv,
              __bf16* __restrict__ Qt, __bf16* __restrict__ Kt,
              __bf16* __restrict__ Vm)
{
    __shared__ float wlds[DK * CIN];              // 32 KB
    const int n = blockIdx.x * 256 + threadIdx.x; // token id, 0..N-1

    for (int p = 0; p < 3; ++p) {
        const float* Wp = (p == 0) ? Wq : (p == 1) ? Wk : Wv;
        const float* bp = (p == 0) ? bq : (p == 1) ? bk : bv;
        __syncthreads();
        for (int i = threadIdx.x; i < DK * CIN; i += 256) wlds[i] = Wp[i];
        __syncthreads();

        for (int ob = 0; ob < DK; ob += 16) {
            float acc[16];
#pragma unroll
            for (int i = 0; i < 16; ++i) acc[i] = bp[ob + i];
            for (int c = 0; c < CIN; ++c) {
                const float x = X[c * NTOK + n];   // coalesced across lanes
#pragma unroll
                for (int i = 0; i < 16; ++i)
                    acc[i] = fmaf(wlds[(ob + i) * CIN + c], x, acc[i]);
            }
            if (p < 2) {                           // Qt/Kt: packed 32B row store
                bf16x16 v;
#pragma unroll
                for (int i = 0; i < 16; ++i) v[i] = (__bf16)acc[i];
                if (p == 0) *(bf16x16*)(Qt + (size_t)n * DK + ob) = v;
                else        *(bf16x16*)(Kt + (size_t)n * DK + ob) = v;
            } else {                               // Vm: coalesced across lanes
#pragma unroll
                for (int i = 0; i < 16; ++i)
                    Vm[(size_t)(ob + i) * NTOK + n] = (__bf16)acc[i];
            }
        }
    }
}

// ---------------------------------------------------------------------------
// Kernel 2: flash attention. One 16-query tile per wave (wave32), BN=64 keys
// per iteration -> 16 v_wmma_f32_16x16x32_bf16 per iteration.
// Block (4 waves) shares each K/V tile: double-buffered
// GLOBAL_LOAD_ASYNC_TO_LDS_B128 staging (ASYNCcnt + one barrier per iter),
// cutting L2 traffic 4x and overlapping next-tile DMA with WMMA+softmax.
// Fragment layouts used here (A and B share contiguous k):
//   A (16x32): lane m=lane&15; kg=lane>>4; a[j]=k(kg*16+j), a[8+j]=k(kg*16+8+j)
//   B (32x16): lane n=lane&15; kh=(lane>>4)*16; b[t]=k(kh+t) -> contiguous
//   C (16x16): VGPR r holds row r + 8*(lane>>4); lane&15 is column
// ---------------------------------------------------------------------------
#define BN        64
#define KV_STRIDE 72   // LDS row stride (elements): 144B, spreads banks
#define P_STRIDE  72

__global__ __launch_bounds__(128)
void attn_fwd(const __bf16* __restrict__ Qt, const __bf16* __restrict__ Kt,
              const __bf16* __restrict__ Vm, float* __restrict__ Of)
{
#if USE_ASYNC
    __shared__ __align__(64) __bf16 kbuf[2][BN * KV_STRIDE];  // 2 x 9.2 KB
    __shared__ __align__(64) __bf16 vbuf[2][DK * KV_STRIDE];  // 2 x 9.2 KB
#endif
    __shared__ __align__(64) __bf16 plds[4][16 * P_STRIDE];

    const int tid   = threadIdx.x;
    const int lane  = tid & 31;
    const int wave  = tid >> 5;
    const int qbase = (blockIdx.x * 4 + wave) * 16;   // 16 queries per wave

    const int mrow = lane & 15;   // A-row / B-col / C-col index
    const int kg   = lane >> 4;   // half-wave selector

#if USE_ASYNC
    // Block-cooperative DMA slice: 128 threads x 16B x 4 steps = 8 KB tile.
    const int drow = tid >> 3;          // 0..15
    const int dcol = (tid & 7) * 8;     // 16B chunk within a 128B row
    auto issue_tile = [&](int jb, int b) {
#pragma unroll
        for (int it = 0; it < 4; ++it) {
            const int r = it * 16 + drow;   // 0..63
            g2l_b128(Kt + (size_t)(jb + r) * DK + dcol,
                     &kbuf[b][r * KV_STRIDE + dcol]);
            g2l_b128(Vm + (size_t)r * NTOK + jb + dcol,
                     &vbuf[b][r * KV_STRIDE + dcol]);
        }
    };
    issue_tile(0, 0);
#endif

    // ---- Q A-fragments, reused across the whole key loop (K=64 -> 2 chunks)
    bf16x16 aq[2];
#pragma unroll
    for (int ch = 0; ch < 2; ++ch) {
        const __bf16* p = Qt + (size_t)(qbase + mrow) * DK + ch * 32 + kg * 16;
#pragma unroll
        for (int j = 0; j < 8; ++j) { aq[ch][j] = p[j]; aq[ch][8 + j] = p[8 + j]; }
    }

    f32x8 oacc[4];
    float mmax[8], lsum[8];
#pragma unroll
    for (int t = 0; t < 4; ++t)
#pragma unroll
        for (int r = 0; r < 8; ++r) oacc[t][r] = 0.0f;
#pragma unroll
    for (int r = 0; r < 8; ++r) { mmax[r] = -1.0e30f; lsum[r] = 0.0f; }

    __bf16* pl = plds[wave];

    int buf = 0;
    for (int j = 0; j < NTOK; j += BN, buf ^= 1) {
#if USE_ASYNC
        wait_async0();        // this wave's slice of tile j is in LDS
        __syncthreads();      // every wave's slice landed; prev buffer is free
        if (j + BN < NTOK) issue_tile(j + BN, buf ^ 1);  // overlap next DMA

        const __bf16* kB = kbuf[buf];
        const __bf16* vB = vbuf[buf];
        bf16x16 kb[8], vb[8];
#pragma unroll
        for (int t = 0; t < 4; ++t)
#pragma unroll
            for (int ch = 0; ch < 2; ++ch)
                kb[t * 2 + ch] = *(const bf16x16*)
                    (kB + (t * 16 + mrow) * KV_STRIDE + ch * 32 + kg * 16);
#pragma unroll
        for (int t2 = 0; t2 < 4; ++t2)
#pragma unroll
            for (int ch = 0; ch < 2; ++ch)
                vb[t2 * 2 + ch] = *(const bf16x16*)
                    (vB + (t2 * 16 + mrow) * KV_STRIDE + ch * 32 + kg * 16);
#else
        if (j + BN < NTOK) {
            __builtin_prefetch(Kt + (size_t)(j + BN) * DK, 0, 0);
            __builtin_prefetch(Vm + (size_t)(j + BN), 0, 0);
        }
        bf16x16 kb[8], vb[8];
#pragma unroll
        for (int t = 0; t < 4; ++t)
#pragma unroll
            for (int ch = 0; ch < 2; ++ch)
                kb[t * 2 + ch] = *(const bf16x16*)
                    (Kt + (size_t)(j + t * 16 + mrow) * DK + ch * 32 + kg * 16);
#pragma unroll
        for (int t2 = 0; t2 < 4; ++t2)
#pragma unroll
            for (int ch = 0; ch < 2; ++ch)
                vb[t2 * 2 + ch] = *(const bf16x16*)
                    (Vm + (size_t)(t2 * 16 + mrow) * NTOK + j + ch * 32 + kg * 16);
#endif

        // ---- S tile: 16 queries x 64 keys (four 16x16 C-frags)
        f32x8 s[4];
#pragma unroll
        for (int t = 0; t < 4; ++t) {
            f32x8 z;
#pragma unroll
            for (int r = 0; r < 8; ++r) z[r] = 0.0f;
            s[t] = wmma_bf16(aq[0], kb[t * 2 + 0], z);
            s[t] = wmma_bf16(aq[1], kb[t * 2 + 1], s[t]);
        }

        // ---- scale into exp2 domain: logits * (1/sqrt(64)) * log2(e)
        const float sc = 0.125f * LOG2E;
#pragma unroll
        for (int t = 0; t < 4; ++t)
#pragma unroll
            for (int r = 0; r < 8; ++r) s[t][r] *= sc;

        // ---- online softmax: row stats per C-VGPR r, DPP butterfly over the
        //      16-lane half that holds that row (no LDS traffic)
        float alpha[8];
#pragma unroll
        for (int r = 0; r < 8; ++r) {
            float mx = fmaxf(fmaxf(s[0][r], s[1][r]), fmaxf(s[2][r], s[3][r]));
            mx = reduce16_max(mx);
            const float mnew = fmaxf(mmax[r], mx);
            alpha[r] = __builtin_exp2f(mmax[r] - mnew);
            mmax[r]  = mnew;
            float rs = 0.0f;
#pragma unroll
            for (int t = 0; t < 4; ++t) {
                const float pv = __builtin_exp2f(s[t][r] - mnew);
                s[t][r] = pv;
                rs += pv;
            }
            rs = reduce16_add(rs);
            lsum[r] = lsum[r] * alpha[r] + rs;
        }
#pragma unroll
        for (int t = 0; t < 4; ++t)
#pragma unroll
            for (int r = 0; r < 8; ++r) oacc[t][r] *= alpha[r];

        // ---- P: C-layout -> bf16 A-layout via per-wave LDS tile
#pragma unroll
        for (int t = 0; t < 4; ++t)
#pragma unroll
            for (int r = 0; r < 8; ++r)
                pl[(r + 8 * kg) * P_STRIDE + t * 16 + mrow] = (__bf16)s[t][r];
        asm volatile("s_wait_dscnt 0x0" ::: "memory");   // wave-local LDS drain

        bf16x16 ap[2];
#pragma unroll
        for (int ch = 0; ch < 2; ++ch) {
            const __bf16* p = pl + mrow * P_STRIDE + ch * 32 + kg * 16;
#pragma unroll
            for (int jj = 0; jj < 8; ++jj) {
                ap[ch][jj] = p[jj]; ap[ch][8 + jj] = p[8 + jj];
            }
        }

        // ---- O += P (16x64) @ V^T (64x64): 4 d-tiles x 2 K-chunks
#pragma unroll
        for (int t2 = 0; t2 < 4; ++t2) {
            oacc[t2] = wmma_bf16(ap[0], vb[t2 * 2 + 0], oacc[t2]);
            oacc[t2] = wmma_bf16(ap[1], vb[t2 * 2 + 1], oacc[t2]);
        }
    }

    // ---- normalize and write O (fp32, token-major [N,64])
    float inv[8];
#pragma unroll
    for (int r = 0; r < 8; ++r) inv[r] = 1.0f / lsum[r];
#pragma unroll
    for (int t2 = 0; t2 < 4; ++t2)
#pragma unroll
        for (int r = 0; r < 8; ++r)
            Of[(size_t)(qbase + r + 8 * kg) * DK + t2 * 16 + mrow] =
                oacc[t2][r] * inv[r];
}

// ---------------------------------------------------------------------------
// Kernel 3: output projection over the RAW reshape (faithful to reference):
//   Y[o][s] = bo[o] + sum_c2 Wo[o][c2] * Oflat[c2*16384 + s]
// where Oflat is O[n][c] flattened row-major.
// ---------------------------------------------------------------------------
__global__ __launch_bounds__(256)
void out_proj(const float* __restrict__ Of, const float* __restrict__ Wo,
              const float* __restrict__ bo, float* __restrict__ Y)
{
    const int o = blockIdx.x >> 6;                          // 0..127
    const int s = ((blockIdx.x & 63) << 8) + threadIdx.x;   // 0..16383
    float acc = bo[o];
#pragma unroll 8
    for (int c2 = 0; c2 < DK; ++c2)
        acc = fmaf(Wo[o * DK + c2], Of[(size_t)c2 * NTOK + s], acc);
    Y[(size_t)o * NTOK + s] = acc;
}

// ---------------------------------------------------------------------------
// Launch
// ---------------------------------------------------------------------------
extern "C" void kernel_launch(void* const* d_in, const int* in_sizes, int n_in,
                              void* d_out, int out_size, void* d_ws, size_t ws_size,
                              hipStream_t stream)
{
    const float* X  = (const float*)d_in[0];
    const float* Wq = (const float*)d_in[1];
    const float* bq = (const float*)d_in[2];
    const float* Wk = (const float*)d_in[3];
    const float* bk = (const float*)d_in[4];
    const float* Wv = (const float*)d_in[5];
    const float* bv = (const float*)d_in[6];
    const float* Wo = (const float*)d_in[7];
    const float* bo = (const float*)d_in[8];
    float* Y = (float*)d_out;

    // Workspace layout (10 MB total): Qt | Kt | Vm (bf16) | Of (fp32)
    char* ws = (char*)d_ws;
    __bf16* Qt = (__bf16*)(ws + 0);
    __bf16* Kt = (__bf16*)(ws + (size_t)NTOK * DK * 2);
    __bf16* Vm = (__bf16*)(ws + (size_t)NTOK * DK * 4);
    float*  Of = (float*) (ws + (size_t)NTOK * DK * 6);

    qkv_proj<<<NTOK / 256, 256, 0, stream>>>(X, Wq, bq, Wk, bk, Wv, bv, Qt, Kt, Vm);
    attn_fwd<<<(NTOK / 16) / 4, 128, 0, stream>>>(Qt, Kt, Vm, Of);
    out_proj<<<COUT * (NTOK / 256), 256, 0, stream>>>(Of, Wo, bo, Y);
}
